// ChunkedLinearDeltaMemory_48928267436326
// MI455X (gfx1250) — matrix operand order repaired and checked by reading, not compiled, not run on hardware
//
#include <hip/hip_runtime.h>

#define D_DIM 384
#define C_CH  64
#define N_CH  128
#define S_LEN 8192
#define B_SZ  8
#define NTHR  768
#define ASTR  392   // padded bf16 row stride for A tile in LDS (384+8)

typedef __attribute__((ext_vector_type(16))) __bf16  v16bf;
typedef __attribute__((ext_vector_type(8)))  float   v8f;

__device__ __forceinline__ unsigned short f2bf(float f) {
  unsigned int u = __float_as_uint(f);
  return (unsigned short)((u + 0x7FFFu + ((u >> 16) & 1u)) >> 16);  // RNE
}
__device__ __forceinline__ unsigned int pk2(float a, float b) {
  return (unsigned int)f2bf(a) | ((unsigned int)f2bf(b) << 16);
}
__device__ __forceinline__ float sigm(float x) { return 1.f / (1.f + __expf(-x)); }

// ---------------------------------------------------------------------------
// Kernel 0: broadcast M_init into f32 master + bf16 mirror, every call.
// ---------------------------------------------------------------------------
__global__ void dm_init_M(const float* __restrict__ Minit,
                          float* __restrict__ Mws,
                          unsigned short* __restrict__ Mbf) {
  int i = blockIdx.x * blockDim.x + threadIdx.x;
  if (i < B_SZ * D_DIM * D_DIM) {
    float v = Minit[i % (D_DIM * D_DIM)];
    Mws[i] = v;
    Mbf[i] = f2bf(v);
  }
}

// ---------------------------------------------------------------------------
// Main kernel: one workgroup per batch, sequential scan over 128 chunks.
// 768 threads = 24 waves; wave w owns output columns [16w, 16w+16).
// All reductions are atomic-free: slot stores + tree/wave reductions.
// ---------------------------------------------------------------------------
__global__ __launch_bounds__(NTHR, 1)
void dm_scan(const float* __restrict__ x,
             const float* __restrict__ eta_raw,
             const float* __restrict__ alpha_raw,
             const float* __restrict__ gate_w,
             const float* __restrict__ gate_b,
             float* __restrict__ out,
             float* __restrict__ Mfin_all,
             float* __restrict__ Mws,
             unsigned short* __restrict__ Mbf_all) {
  __shared__ unsigned short Abf[C_CH * ASTR];   // chunk as bf16, padded rows (50176 B)
  __shared__ float sm[2393];                    // reused scratch + finals (9572 B)
  float* s0  = sm;            // 768: rnorm partials (P1a)  -> v_mean partials (P2)
  float* s1  = sm + 768;      // 768: k_mean partials (P1c) -> Mk partials (P3)
  float* rn  = sm + 1536;     // 64 : 1/max(rownorm,eps)
  float* km  = sm + 1600;     // 384: k_mean (final)
  float* vm  = sm + 1984;     // 384: v_mean (final)
  float* nsw = sm + 2368;     // 24 : per-wave ||M||^2 partials
  const int GD = 2392;        // gate dot

  const int b    = blockIdx.x;
  const int tid  = threadIdx.x;
  const int lane = tid & 31;
  const int w    = tid >> 5;       // wave id 0..23
  const bool hi  = lane >= 16;
  const int l16  = lane & 15;

  const float eta   = 0.2f * sigm(eta_raw[0]);
  const float alpha = 0.5f + 0.5f * sigm(alpha_raw[0]);
  const float gb    = gate_b[0];

  const float* xb = x   + (size_t)b * S_LEN * D_DIM;
  float*       ob = out + (size_t)b * S_LEN * D_DIM;
  float*       M  = Mws + (size_t)b * D_DIM * D_DIM;
  unsigned short* Mbf = Mbf_all + (size_t)b * D_DIM * D_DIM;
  float*     Mfin = Mfin_all + (size_t)b * D_DIM * D_DIM;

  for (int step = 0; step < N_CH; ++step) {
    const float* xc = xb + (size_t)step * C_CH * D_DIM;
    float*       oc = ob + (size_t)step * C_CH * D_DIM;

    // ---- P1a: load chunk, row sum-of-squares partials, stage bf16 A -----
    {
      const int r = tid / 12;          // row 0..63
      const int j = tid % 12;          // 32-elem segment
      const float4* src = (const float4*)(xc + r * D_DIM + j * 32);
      float s = 0.f;
      unsigned int pk[16];
#pragma unroll
      for (int i = 0; i < 8; ++i) {
        float4 v = src[i];
        s += v.x*v.x + v.y*v.y + v.z*v.z + v.w*v.w;
        pk[2*i]   = pk2(v.x, v.y);
        pk[2*i+1] = pk2(v.z, v.w);
      }
      uint4* dq = (uint4*)&Abf[r * ASTR + j * 32];
#pragma unroll
      for (int q = 0; q < 4; ++q)
        dq[q] = make_uint4(pk[4*q], pk[4*q+1], pk[4*q+2], pk[4*q+3]);
      s0[r * 12 + j] = s;              // partial, no atomic
    }
    __syncthreads();

    // ---- P1b: inverse row norms (F.normalize eps = 1e-5) ----------------
    if (tid < C_CH) {
      float s = 0.f;
#pragma unroll
      for (int j = 0; j < 12; ++j) s += s0[tid * 12 + j];
      rn[tid] = 1.f / fmaxf(sqrtf(s), 1e-5f);
    }
    __syncthreads();

    // ---- P1c: k_mean partials (f32 from global, coalesced) --------------
    {
      const int d = tid % D_DIM;
      const int h = tid / D_DIM;               // 0 or 1 -> rows 32h..32h+31
      const float* src = xc + (size_t)h * 32 * D_DIM + d;
      float s = 0.f;
#pragma unroll 8
      for (int c = 0; c < 32; ++c) s += src[c * D_DIM] * rn[32*h + c];
      s1[h * D_DIM + d] = s;           // partial, no atomic
    }
    __syncthreads();
    // finalize k_mean
    if (tid < D_DIM) km[tid] = (s1[tid] + s1[D_DIM + tid]) * (1.f / 64.f);
    __syncthreads();

    // ---- P2: retrieval GEMM  out = chunk @ M^T  via bf16 WMMA -----------
    // Wave 0 additionally computes the gate dot via shuffle reduction.
    if (w == 0) {
      float g = 0.f;
#pragma unroll
      for (int q = 0; q < 12; ++q) {
        const int d = lane + 32 * q;
        g += km[d] * gate_w[d];
      }
#pragma unroll
      for (int off = 16; off; off >>= 1) g += __shfl_xor(g, off, 32);
      if (lane == 0) sm[GD] = g;
    }
    {
      // prefetch next chunk of x into near caches while WMMAs run
      if (step + 1 < N_CH) {
        const char* nx = (const char*)(xc + (size_t)C_CH * D_DIM);
        __builtin_prefetch(nx + (size_t)tid * 128, 0, 3);
      }

      const int obase = w * 16;
      const unsigned short* Brow = Mbf + (size_t)(obase + l16) * D_DIM + (hi ? 8 : 0);
      const unsigned short* Arow = Abf + l16 * ASTR + (hi ? 8 : 0);
      v8f acc[4];
#pragma unroll
      for (int t = 0; t < 4; ++t) acc[t] = (v8f){0,0,0,0,0,0,0,0};

#pragma unroll
      for (int kk = 0; kk < 12; ++kk) {
        const int kb = kk * 32;
        // Issue B (global) and ALL four A tiles (LDS) before any WMMA so the
        // dependency waits are progressive instead of dscnt==0 per tile.
        union { uint4 q[2]; v16bf v; } B, A0, A1, A2, A3;
        const uint4* bp = (const uint4*)(Brow + kb);
        B.q[0] = bp[0];
        B.q[1] = bp[2];
        const unsigned short* ap0 = Arow + kb;
        A0.q[0] = *(const uint4*)ap0;
        A0.q[1] = *(const uint4*)(ap0 + 16);
        const unsigned short* ap1 = ap0 + 16 * ASTR;
        A1.q[0] = *(const uint4*)ap1;
        A1.q[1] = *(const uint4*)(ap1 + 16);
        const unsigned short* ap2 = ap1 + 16 * ASTR;
        A2.q[0] = *(const uint4*)ap2;
        A2.q[1] = *(const uint4*)(ap2 + 16);
        const unsigned short* ap3 = ap2 + 16 * ASTR;
        A3.q[0] = *(const uint4*)ap3;
        A3.q[1] = *(const uint4*)(ap3 + 16);

        acc[0] = __builtin_amdgcn_wmma_f32_16x16x32_bf16(
            false, A0.v, false, B.v, (short)0, acc[0], false, false);
        acc[1] = __builtin_amdgcn_wmma_f32_16x16x32_bf16(
            false, A1.v, false, B.v, (short)0, acc[1], false, false);
        acc[2] = __builtin_amdgcn_wmma_f32_16x16x32_bf16(
            false, A2.v, false, B.v, (short)0, acc[2], false, false);
        acc[3] = __builtin_amdgcn_wmma_f32_16x16x32_bf16(
            false, A3.v, false, B.v, (short)0, acc[3], false, false);
      }

      // store out + per-half-wave column sums for v_mean (slot, no atomic)
      const int ocol = obase + l16;
      float vs = 0.f;
#pragma unroll
      for (int tm = 0; tm < 4; ++tm) {
        const int crow = tm * 16 + (hi ? 8 : 0);
#pragma unroll
        for (int r = 0; r < 8; ++r) {
          float v = acc[tm][r];
          oc[(size_t)(crow + r) * D_DIM + ocol] = v;
          vs += v;
        }
      }
      s0[(hi ? D_DIM : 0) + ocol] = vs;
    }
    __syncthreads();

    // ---- finalize v_mean; P3: Mk partials (f32 master) ------------------
    if (tid < D_DIM) vm[tid] = (s0[tid] + s0[D_DIM + tid]) * (1.f / 64.f);
    {
      const int o = tid >> 1;
      const int h = tid & 1;
      const float* mr  = M + (size_t)o * D_DIM + h * 192;
      const float* kmr = km + h * 192;
      float s = 0.f;
#pragma unroll 8
      for (int d = 0; d < 192; ++d) s += mr[d] * kmr[d];
      s1[h * D_DIM + o] = s;           // Mk partials, no atomic
    }
    __syncthreads();

    // ---- P4a: delta update + gate blend, accumulate ||M_new||^2 ---------
    // Pair mapping: each thread owns d-pairs (d0, d0+1), o advancing by 4.
    const int d0 = 2 * (tid % 192);
    const int o0 = tid / 192;                    // 0..3
    {
      const float gate = sigm(sm[GD] + gb);
      const float cm = gate * alpha + (1.f - gate);   // coeff on old M
      const float ge = gate * eta;                    // coeff on update
      const float k0 = km[d0], k1 = km[d0 + 1];
      float ns = 0.f;
#pragma unroll 4
      for (int j = 0; j < 96; ++j) {
        const int o = o0 + 4 * j;
        const int i = o * D_DIM + d0;
        float2 m = *(const float2*)(M + i);
        const float e = ge * (vm[o] - (s1[o] + s1[D_DIM + o]));   // Mk inline
        float n0 = cm * m.x + e * k0;
        float n1 = cm * m.y + e * k1;
        *(float2*)(M + i) = make_float2(n0, n1);
        ns += n0 * n0 + n1 * n1;
      }
#pragma unroll
      for (int off = 16; off; off >>= 1) ns += __shfl_xor(ns, off, 32);
      if (lane == 0) nsw[w] = ns;      // per-wave slot, no atomic
    }
    __syncthreads();

    // ---- P4b: Frobenius clamp + refresh bf16 mirror (own elements) ------
    {
      float ns2 = 0.f;
#pragma unroll
      for (int q = 0; q < 24; ++q) ns2 += nsw[q];   // same order for all -> deterministic
      const float nrm = sqrtf(ns2);
      const float scl = fminf(30.f / (nrm + 1e-6f), 1.f);
      const bool last = (step == N_CH - 1);
#pragma unroll 4
      for (int j = 0; j < 96; ++j) {
        const int i = (o0 + 4 * j) * D_DIM + d0;
        float2 m = *(const float2*)(M + i);
        float v0 = m.x * scl, v1 = m.y * scl;
        *(float2*)(M + i) = make_float2(v0, v1);
        ((unsigned int*)Mbf)[i >> 1] = pk2(v0, v1);   // bf16 mirror for next GEMM
        if (last) *(float2*)(Mfin + i) = make_float2(v0, v1);
      }
    }
    __threadfence();      // publish M / Mbf to the whole workgroup
    __syncthreads();
  }
}

// ---------------------------------------------------------------------------
extern "C" void kernel_launch(void* const* d_in, const int* in_sizes, int n_in,
                              void* d_out, int out_size, void* d_ws, size_t ws_size,
                              hipStream_t stream) {
  const float* x    = (const float*)d_in[0];
  const float* Mi   = (const float*)d_in[1];
  const float* er   = (const float*)d_in[2];
  const float* ar   = (const float*)d_in[3];
  const float* gw   = (const float*)d_in[4];
  const float* gbp  = (const float*)d_in[5];
  float* outp = (float*)d_out;
  float* Mws  = (float*)d_ws;                                  // 8*384*384 f32
  unsigned short* Mbf = (unsigned short*)(Mws + (size_t)B_SZ * D_DIM * D_DIM);
  float* Mfin = outp + (size_t)B_SZ * S_LEN * D_DIM;           // outputs then M_fin

  const int mtot = B_SZ * D_DIM * D_DIM;
  dm_init_M<<<(mtot + 255) / 256, 256, 0, stream>>>(Mi, Mws, Mbf);
  dm_scan<<<B_SZ, NTHR, 0, stream>>>(x, er, ar, gw, gbp, outp, Mfin, Mws, Mbf);
}
